// Rank1EditModule_6433861009600
// MI455X (gfx1250) — compile-verified, matrix operand order checked by reading, not run on hardware
//
#include <hip/hip_runtime.h>
#include <cstdint>

#define NPROMPT 512
#define SEQ 256
#define DIM 1024
#define BATCH 64
#define BETA 0.75f
#define TEMP 0.1f

typedef __attribute__((ext_vector_type(16))) __bf16 v16bf;
typedef __attribute__((ext_vector_type(8)))  float v8f;

union FragU { unsigned int u[8]; v16bf v; };

__device__ __forceinline__ unsigned int pack_bf16(float a, float b) {
#if __has_builtin(__builtin_amdgcn_cvt_pk_bf16_f32)
  auto r = __builtin_amdgcn_cvt_pk_bf16_f32(a, b);
  unsigned int u;
  __builtin_memcpy(&u, &r, sizeof(u));
  return u;
#else
  // round-to-nearest-even f32 -> bf16, packed pair (lo = a, hi = b)
  unsigned int ua = __float_as_uint(a);
  unsigned int ub = __float_as_uint(b);
  ua = (ua + 0x7FFFu + ((ua >> 16) & 1u)) >> 16;
  ub = (ub + 0x7FFFu + ((ub >> 16) & 1u)) >> 16;
  return (ub << 16) | (ua & 0xFFFFu);
#endif
}

__device__ __forceinline__ uint4 pack_row8(float4 x0, float4 x1) {
  return make_uint4(pack_bf16(x0.x, x0.y), pack_bf16(x0.z, x0.w),
                    pack_bf16(x1.x, x1.y), pack_bf16(x1.z, x1.w));
}

// ---------------------------------------------------------------------------
// 1) ema_concept[b,:] = initted[pid] ? ema_table[pid,:] : text_enc[b, ci[b], :]
// ---------------------------------------------------------------------------
__global__ __launch_bounds__(256) void k_prep_ema(
    const int* __restrict__ prompt_ids, const int* __restrict__ concept_idx,
    const unsigned char* __restrict__ initted,
    const float* __restrict__ text_enc, const float* __restrict__ ema_table,
    float* __restrict__ e) {
  int b = blockIdx.x;
  int d0 = threadIdx.x * 4;
  int pid = prompt_ids[b];
  bool init = initted[pid] != 0;
  const float4* src = init
      ? (const float4*)(ema_table + (size_t)pid * DIM + d0)
      : (const float4*)(text_enc + ((size_t)b * SEQ + concept_idx[b]) * DIM + d0);
  *(float4*)(e + (size_t)b * DIM + d0) = *src;
}

// ---------------------------------------------------------------------------
// 2) out(64x1024) = A(64x1024) @ Bm(1024x1024)^T   -- bf16 WMMA, f32 accum
//    Block tile 64(M) x 128(N); 8 waves = 4(M) x 2(N); each wave 16x64.
// ---------------------------------------------------------------------------
__global__ __launch_bounds__(256) void k_small_gemm64(
    const float* __restrict__ A, const float* __restrict__ Bm,
    float* __restrict__ out) {
  __shared__ unsigned int As[64 * 16];    // 64 rows x 32 bf16 (16 uints)
  __shared__ unsigned int Bs[128 * 16];   // 128 rows x 32 bf16
  int t = threadIdx.x;
  int lane = t & 31, wave = t >> 5;
  int wm = wave >> 1, wn = wave & 1;
  int block_o = blockIdx.x * 128;
  int ar = t >> 2, kg = t & 3;

  v8f acc[4] = {};

  const float* Aptr = A + (size_t)ar * DIM + kg * 8;
  const float* Bp0 = Bm + (size_t)(block_o + ar) * DIM + kg * 8;
  const float* Bp1 = Bm + (size_t)(block_o + ar + 64) * DIM + kg * 8;

  for (int k0 = 0; k0 < DIM; k0 += 32) {
    float4 x0 = *(const float4*)(Aptr + k0);
    float4 x1 = *(const float4*)(Aptr + k0 + 4);
    float4 y0 = *(const float4*)(Bp0 + k0);
    float4 y1 = *(const float4*)(Bp0 + k0 + 4);
    float4 z0 = *(const float4*)(Bp1 + k0);
    float4 z1 = *(const float4*)(Bp1 + k0 + 4);
    *(uint4*)(&As[ar * 16 + kg * 4]) = pack_row8(x0, x1);
    *(uint4*)(&Bs[ar * 16 + kg * 4]) = pack_row8(y0, y1);
    *(uint4*)(&Bs[(ar + 64) * 16 + kg * 4]) = pack_row8(z0, z1);
    if (k0 + 32 < DIM) {
      __builtin_prefetch(Aptr + k0 + 32, 0, 3);
      __builtin_prefetch(Bp0 + k0 + 32, 0, 3);
      __builtin_prefetch(Bp1 + k0 + 32, 0, 3);
    }
    __syncthreads();

    // Fragment: lane L holds row M=L%16; K chunks [kb..kb+7],[kb+16..kb+23], kb=8*(L/16)
    int khalf = (lane >> 4) * 4;               // uint offset of first chunk
    int arow = wm * 16 + (lane & 15);
    FragU fa;
    {
      const uint4* p = (const uint4*)(&As[arow * 16 + khalf]);
      *(uint4*)(&fa.u[0]) = p[0];
      *(uint4*)(&fa.u[4]) = p[2];
    }
#pragma unroll
    for (int tt = 0; tt < 4; ++tt) {
      int brow = wn * 64 + tt * 16 + (lane & 15);
      FragU fb;
      const uint4* p = (const uint4*)(&Bs[brow * 16 + khalf]);
      *(uint4*)(&fb.u[0]) = p[0];
      *(uint4*)(&fb.u[4]) = p[2];
      acc[tt] = __builtin_amdgcn_wmma_f32_16x16x32_bf16(
          false, fa.v, false, fb.v, (short)0, acc[tt], false, false);
    }
    __syncthreads();
  }

#pragma unroll
  for (int tt = 0; tt < 4; ++tt)
#pragma unroll
    for (int r = 0; r < 8; ++r) {
      int m = wm * 16 + r + ((lane >> 4) << 3);
      int o = block_o + wn * 64 + tt * 16 + (lane & 15);
      out[(size_t)m * DIM + o] = acc[tt][r];
    }
}

// ---------------------------------------------------------------------------
// 3) i_energy[b] = dot(iCi[b,:], e[b,:])
// ---------------------------------------------------------------------------
__global__ __launch_bounds__(256) void k_energy(
    const float* __restrict__ iCi, const float* __restrict__ e,
    float* __restrict__ energy) {
  __shared__ float red[256];
  int b = blockIdx.x;
  int d0 = threadIdx.x * 4;
  float4 x = *(const float4*)(iCi + (size_t)b * DIM + d0);
  float4 y = *(const float4*)(e + (size_t)b * DIM + d0);
  red[threadIdx.x] = x.x * y.x + x.y * y.y + x.z * y.z + x.w * y.w;
  __syncthreads();
  for (int s = 128; s > 0; s >>= 1) {
    if (threadIdx.x < s) red[threadIdx.x] += red[threadIdx.x + s];
    __syncthreads();
  }
  if (threadIdx.x == 0) energy[b] = red[0];
}

// ---------------------------------------------------------------------------
// 4) per-row: sim = <text_enc[m], iCi[b]>; alpha = sim/E; sig = sigmoid((alpha-B)/T)
//    one wave32 per row m
// ---------------------------------------------------------------------------
__global__ __launch_bounds__(256) void k_sim(
    const float* __restrict__ text_enc, const float* __restrict__ iCi,
    const float* __restrict__ energy,
    float* __restrict__ alpha, float* __restrict__ sig) {
  int lane = threadIdx.x & 31;
  int wave = threadIdx.x >> 5;
  int m = blockIdx.x * 8 + wave;
  int b = m >> 8;
  const float* te = text_enc + (size_t)m * DIM;
  const float* ic = iCi + (size_t)b * DIM;
  float acc = 0.f;
#pragma unroll
  for (int k = lane * 4; k < DIM; k += 128) {
    float4 x = *(const float4*)(te + k);
    float4 y = *(const float4*)(ic + k);
    acc += x.x * y.x + x.y * y.y + x.z * y.z + x.w * y.w;
  }
#pragma unroll
  for (int off = 16; off > 0; off >>= 1) acc += __shfl_down(acc, off, 32);
  if (lane == 0) {
    float a = acc / energy[b];
    alpha[m] = a;
    sig[m] = 1.0f / (1.0f + __expf(-(a - BETA) / TEMP));
  }
}

// ---------------------------------------------------------------------------
// 5) Main fused GEMM + epilogue:
//    orig = text_enc @ weight^T  (bf16 WMMA, f32 accum)
//    out  = orig - alpha*co + sig * (initted ? outputs_table : orig)
//    Block tile 128(M) x 128(N); 8 waves = 4(M) x 2(N); each wave 32x64
//    -> 8 WMMAs per wave per K-step (2 A-frags x 4 B-frags).
// ---------------------------------------------------------------------------
__global__ __launch_bounds__(256) void k_main_gemm(
    const float* __restrict__ text_enc, const float* __restrict__ weight,
    const float* __restrict__ alpha, const float* __restrict__ sig,
    const float* __restrict__ co, const float* __restrict__ outputs_table,
    const int* __restrict__ prompt_ids, const unsigned char* __restrict__ initted,
    float* __restrict__ out) {
  __shared__ unsigned int As[128 * 16];   // 128 rows x 32 bf16
  __shared__ unsigned int Bs[128 * 16];   // 128 rows x 32 bf16
  int t = threadIdx.x;
  int lane = t & 31, wave = t >> 5;
  int wm = wave >> 1, wn = wave & 1;      // wave tile: rows wm*32..+31, cols wn*64..+63
  int block_m = blockIdx.x * 128;
  int block_o = blockIdx.y * 128;
  int ar = t >> 2, kg = t & 3;            // each thread: 2 A row-chunks + 2 B row-chunks

  v8f acc[2][4] = {};

  const float* ApA0 = text_enc + (size_t)(block_m + ar) * DIM + kg * 8;
  const float* ApA1 = text_enc + (size_t)(block_m + ar + 64) * DIM + kg * 8;
  const float* BpB0 = weight + (size_t)(block_o + ar) * DIM + kg * 8;
  const float* BpB1 = weight + (size_t)(block_o + ar + 64) * DIM + kg * 8;

  for (int k0 = 0; k0 < DIM; k0 += 32) {
    float4 a00 = *(const float4*)(ApA0 + k0);
    float4 a01 = *(const float4*)(ApA0 + k0 + 4);
    float4 a10 = *(const float4*)(ApA1 + k0);
    float4 a11 = *(const float4*)(ApA1 + k0 + 4);
    float4 b00 = *(const float4*)(BpB0 + k0);
    float4 b01 = *(const float4*)(BpB0 + k0 + 4);
    float4 b10 = *(const float4*)(BpB1 + k0);
    float4 b11 = *(const float4*)(BpB1 + k0 + 4);
    *(uint4*)(&As[ar * 16 + kg * 4]) = pack_row8(a00, a01);
    *(uint4*)(&As[(ar + 64) * 16 + kg * 4]) = pack_row8(a10, a11);
    *(uint4*)(&Bs[ar * 16 + kg * 4]) = pack_row8(b00, b01);
    *(uint4*)(&Bs[(ar + 64) * 16 + kg * 4]) = pack_row8(b10, b11);
    if (k0 + 32 < DIM) {
      __builtin_prefetch(ApA0 + k0 + 32, 0, 3);
      __builtin_prefetch(ApA1 + k0 + 32, 0, 3);
      __builtin_prefetch(BpB0 + k0 + 32, 0, 3);
      __builtin_prefetch(BpB1 + k0 + 32, 0, 3);
    }
    __syncthreads();

    int khalf = (lane >> 4) * 4;
    FragU fa[2];
#pragma unroll
    for (int am = 0; am < 2; ++am) {
      int arow = wm * 32 + am * 16 + (lane & 15);
      const uint4* p = (const uint4*)(&As[arow * 16 + khalf]);
      *(uint4*)(&fa[am].u[0]) = p[0];
      *(uint4*)(&fa[am].u[4]) = p[2];
    }
#pragma unroll
    for (int tt = 0; tt < 4; ++tt) {
      int brow = wn * 64 + tt * 16 + (lane & 15);
      FragU fb;
      const uint4* p = (const uint4*)(&Bs[brow * 16 + khalf]);
      *(uint4*)(&fb.u[0]) = p[0];
      *(uint4*)(&fb.u[4]) = p[2];
#pragma unroll
      for (int am = 0; am < 2; ++am)
        acc[am][tt] = __builtin_amdgcn_wmma_f32_16x16x32_bf16(
            false, fa[am].v, false, fb.v, (short)0, acc[am][tt], false, false);
    }
    __syncthreads();
  }

  // Fused epilogue. C/D layout: VGPR r, lane L -> M = r + 8*(L/16), N = L%16.
#pragma unroll
  for (int am = 0; am < 2; ++am)
#pragma unroll
    for (int tt = 0; tt < 4; ++tt)
#pragma unroll
      for (int r = 0; r < 8; ++r) {
        int m = block_m + wm * 32 + am * 16 + r + ((lane >> 4) << 3);
        int o = block_o + wn * 64 + tt * 16 + (lane & 15);
        int b = m >> 8;
        int n = m & (SEQ - 1);
        int pid = prompt_ids[b];
        bool init = initted[pid] != 0;
        float orig = acc[am][tt][r];
        float al = alpha[m];
        float sg = sig[m];
        float cv = co[(size_t)b * DIM + o];
        float ov = init ? outputs_table[((size_t)pid * SEQ + n) * DIM + o] : orig;
        out[(size_t)m * DIM + o] = orig - al * cv + sg * ov;
      }
}

// ---------------------------------------------------------------------------
extern "C" void kernel_launch(void* const* d_in, const int* in_sizes, int n_in,
                              void* d_out, int out_size, void* d_ws, size_t ws_size,
                              hipStream_t stream) {
  (void)in_sizes; (void)n_in; (void)out_size; (void)ws_size;
  const int* prompt_ids = (const int*)d_in[0];
  const float* text_enc = (const float*)d_in[1];
  const int* concept_idx = (const int*)d_in[2];
  const float* weight = (const float*)d_in[3];
  const float* C_inv = (const float*)d_in[4];
  const unsigned char* initted = (const unsigned char*)d_in[5];
  const float* ema_table = (const float*)d_in[6];
  const float* outputs_table = (const float*)d_in[7];
  float* out = (float*)d_out;

  float* ws = (float*)d_ws;
  float* e = ws;                       // 64*1024
  float* iCi = e + BATCH * DIM;        // 64*1024
  float* co = iCi + BATCH * DIM;       // 64*1024
  float* energy = co + BATCH * DIM;    // 64
  float* alpha = energy + BATCH;       // 64*256
  float* sig = alpha + BATCH * SEQ;    // 64*256

  k_prep_ema<<<BATCH, 256, 0, stream>>>(prompt_ids, concept_idx, initted,
                                        text_enc, ema_table, e);
  // iCi = e @ C_inv  (C_inv symmetric => e @ C_inv^T identical, row-contig B reads)
  k_small_gemm64<<<DIM / 128, 256, 0, stream>>>(e, C_inv, iCi);
  // co = e @ weight^T
  k_small_gemm64<<<DIM / 128, 256, 0, stream>>>(e, weight, co);
  k_energy<<<BATCH, 256, 0, stream>>>(iCi, e, energy);
  k_sim<<<(BATCH * SEQ) / 8, 256, 0, stream>>>(text_enc, iCi, energy, alpha, sig);
  dim3 grid((BATCH * SEQ) / 128, DIM / 128);
  k_main_gemm<<<grid, 256, 0, stream>>>(text_enc, weight, alpha, sig, co,
                                        outputs_table, prompt_ids, initted, out);
}
